// LightEncoderLayer_7619271983553
// MI455X (gfx1250) — matrix operand order repaired and checked by reading, not compiled, not run on hardware
//
#include <hip/hip_runtime.h>
#include <hip/hip_bf16.h>

typedef __bf16 v16bf __attribute__((ext_vector_type(16)));
typedef __bf16 v2bf  __attribute__((ext_vector_type(2)));
typedef float  v8f   __attribute__((ext_vector_type(8)));

#define B_  128
#define S_  512
#define H_  16
#define D_  1024
#define DI_ 512
#define EP_ 64

union FragBF { uint4 q[2]; unsigned u[8]; v16bf v; };

#if __has_builtin(__builtin_amdgcn_cvt_pk_bf16_f32)
__device__ __forceinline__ unsigned pack_bf16(float a, float b) {
  union { v2bf v; unsigned u; } cv;
  cv.v = __builtin_amdgcn_cvt_pk_bf16_f32(a, b);   // v_cvt_pk_bf16_f32
  return cv.u;
}
#else
__device__ __forceinline__ unsigned pack_bf16(float a, float b) {
  unsigned ua = __float_as_uint(a), ub = __float_as_uint(b);
  ua = ua + 0x7FFFu + ((ua >> 16) & 1u);   // round-to-nearest-even
  ub = ub + 0x7FFFu + ((ub >> 16) & 1u);
  return (ua >> 16) | (ub & 0xFFFF0000u);
}
#endif
__device__ __forceinline__ unsigned short bf16_1(float a) {
  unsigned ua = __float_as_uint(a);
  ua = ua + 0x7FFFu + ((ua >> 16) & 1u);
  return (unsigned short)(ua >> 16);
}
__device__ __forceinline__ void wave_lds_fence() {
  asm volatile("s_wait_dscnt 0x0" ::: "memory");
}

// ---------------------------------------------------------------------------
// Kernel 1: k = x @ Wk + bk (bf16 WMMA) fused with scores = (Q.k)/sqrt(8).
// WG: 256 threads (8 waves), 128 tokens x 128 cols.
// Double-buffered LDS ping-pong (A0,B0,A1,B1) + register prefetch.
// All global loads are b128. Dyn LDS = 96KB.
// ---------------------------------------------------------------------------
struct TileRegs { float4 xv[4]; float4 w0[2]; float4 w1[2]; };

__device__ __forceinline__ void fetch_tile(const float* __restrict__ x,
                                           const float* __restrict__ Wk,
                                           int T0, int k0, int t, TileRegs& r) {
#pragma unroll
  for (int c = 0; c < 4; ++c) {
    int f4i = t + c * 256;
    int row = f4i >> 3, f4 = f4i & 7;
    r.xv[c] = *(const float4*)(x + (size_t)(T0 + row) * D_ + k0 + f4 * 4);
  }
  // Wk tile: each thread covers 2 groups of (kpair, 4 consecutive n)
#pragma unroll
  for (int g = 0; g < 2; ++g) {
    int gidx = t + g * 256;
    int kp = gidx >> 5, n0 = (gidx & 31) * 4;
    r.w0[g] = *(const float4*)(Wk + (size_t)(k0 + 2 * kp) * 128 + n0);
    r.w1[g] = *(const float4*)(Wk + (size_t)(k0 + 2 * kp + 1) * 128 + n0);
  }
}
__device__ __forceinline__ void store_tile(unsigned* bufA, unsigned* bufB,
                                           int t, const TileRegs& r) {
#pragma unroll
  for (int c = 0; c < 4; ++c) {
    int f4i = t + c * 256;
    int row = f4i >> 3, f4 = f4i & 7;
    uint2 pk;
    pk.x = pack_bf16(r.xv[c].x, r.xv[c].y);
    pk.y = pack_bf16(r.xv[c].z, r.xv[c].w);
    *(uint2*)(bufA + row * 16 + f4 * 2) = pk;       // ds_store_b64
  }
#pragma unroll
  for (int g = 0; g < 2; ++g) {
    int gidx = t + g * 256;
    int kp = gidx >> 5, n0 = (gidx & 31) * 4;
    bufB[(n0 + 0) * 16 + kp] = pack_bf16(r.w0[g].x, r.w1[g].x);
    bufB[(n0 + 1) * 16 + kp] = pack_bf16(r.w0[g].y, r.w1[g].y);
    bufB[(n0 + 2) * 16 + kp] = pack_bf16(r.w0[g].z, r.w1[g].z);
    bufB[(n0 + 3) * 16 + kp] = pack_bf16(r.w0[g].w, r.w1[g].w);
  }
}

__global__ __launch_bounds__(256) void kproj_scores_kernel(
    const float* __restrict__ x, const float* __restrict__ Wk,
    const float* __restrict__ bk, const float* __restrict__ Q,
    float* __restrict__ scores)
{
  extern __shared__ unsigned smem[];
  // layout (dwords): A0 @0, B0 @2048, A1 @4096, B1 @6144, sC @8192
  float* sC = (float*)(smem + 8192);                // 128 x 128 f32

  const int t = threadIdx.x;
  const int w = t >> 5, l = t & 31;
  const int half = l >> 4, l15 = l & 15;
  const int T0 = blockIdx.x * 128;

  v8f acc[8];
#pragma unroll
  for (int nt = 0; nt < 8; ++nt)
#pragma unroll
    for (int i = 0; i < 8; ++i) acc[nt][i] = 0.f;

  TileRegs r;
  fetch_tile(x, Wk, T0, 0, t, r);
  store_tile(smem, smem + 2048, t, r);
  fetch_tile(x, Wk, T0, 32, t, r);
  __syncthreads();

  for (int it = 0; it < 32; ++it) {
    unsigned* bA = smem + (it & 1) * 4096;
    unsigned* bB = bA + 2048;

    // A fragment: lane<16 -> K pairs {0..3, 8..11}; lane>=16 -> {4..7, 12..15}
    FragBF af;
    const unsigned* pA = bA + (w * 16 + l15) * 16 + half * 4;
    af.q[0] = *(const uint4*)pA;                    // ds_load_b128
    af.q[1] = *(const uint4*)(pA + 8);
#pragma unroll
    for (int nt = 0; nt < 8; ++nt) {
      FragBF bfv;
      const unsigned* pB = bB + (nt * 16 + l15) * 16 + half * 4;
      bfv.q[0] = *(const uint4*)pB;
      bfv.q[1] = *(const uint4*)(pB + 8);
      acc[nt] = __builtin_amdgcn_wmma_f32_16x16x32_bf16(
          false, af.v, false, bfv.v, (short)0, acc[nt], false, false);
    }
    if (it < 31) {
      __syncthreads();                              // all waves done reading buf[1-cur]
      unsigned* nA = smem + ((it + 1) & 1) * 4096;
      store_tile(nA, nA + 2048, t, r);
      if (it < 30) fetch_tile(x, Wk, T0, (it + 2) * 32, t, r);
      __syncthreads();                              // stores visible before next reads
    }
  }

  // bias + spill C to LDS (k never goes to HBM)
#pragma unroll
  for (int nt = 0; nt < 8; ++nt) {
    int n = nt * 16 + l15;
    float bias = bk[n];
#pragma unroll
    for (int i = 0; i < 8; ++i) {
      int m = i + 8 * half;
      sC[(w * 16 + m) * 128 + n] = acc[nt][i] + bias;
    }
  }
  __syncthreads();

  // scores[n = h*128 + b][s] = Q[h] . k[token][h*8..] * rsqrt(8)
  {
    int h = t >> 4;
    float q[8];
#pragma unroll
    for (int d = 0; d < 8; ++d) q[d] = Q[h * 8 + d];
#pragma unroll
    for (int j = 0; j < 8; ++j) {
      int row = j * 16 + (t & 15);
      const float4* kp = (const float4*)(sC + row * 128 + h * 8);
      float4 k0 = kp[0], k1 = kp[1];
      float s = q[0]*k0.x + q[1]*k0.y + q[2]*k0.z + q[3]*k0.w
              + q[4]*k1.x + q[5]*k1.y + q[6]*k1.z + q[7]*k1.w;
      int T = T0 + row;
      int b = T >> 9, sp = T & 511;
      scores[(size_t)(h * B_ + b) * S_ + sp] = s * 0.35355339059327373f;
    }
  }
}

// ---------------------------------------------------------------------------
// Kernel 2: masked row softmax over flat 2048 x 512; P in place; the
// attn_weight output IS this flat array (attn_weight[b,h,s] = P[b*16+h][s]).
// ---------------------------------------------------------------------------
__global__ __launch_bounds__(256) void softmax_kernel(
    float* __restrict__ P, const unsigned char* __restrict__ mask,
    float* __restrict__ attn_w)
{
  __shared__ float red[256];
  const int r = blockIdx.x, t = threadIdx.x;
  const size_t base = (size_t)r * S_;
  float v0 = P[base + t];
  float v1 = P[base + t + 256];
  if (mask[base + t])       v0 = -1e8f;
  if (mask[base + t + 256]) v1 = -1e8f;
  float m = fmaxf(v0, v1);
  red[t] = m; __syncthreads();
#pragma unroll
  for (int o = 128; o > 0; o >>= 1) {
    if (t < o) red[t] = fmaxf(red[t], red[t + o]);
    __syncthreads();
  }
  m = red[0]; __syncthreads();
  float e0 = __expf(v0 - m), e1 = __expf(v1 - m);
  red[t] = e0 + e1; __syncthreads();
#pragma unroll
  for (int o = 128; o > 0; o >>= 1) {
    if (t < o) red[t] += red[t + o];
    __syncthreads();
  }
  float inv = 1.f / red[0];
  float p0 = e0 * inv, p1 = e1 * inv;
  P[base + t] = p0;          P[base + t + 256] = p1;
  attn_w[base + t] = p0;     attn_w[base + t + 256] = p1;
}

// ---------------------------------------------------------------------------
// Kernel 3: head mixture + LN1 + FFN(64->512->64, bf16 WMMA) + residual + LN2.
// WG: 256 threads (8 waves), 128 tokens. Dyn LDS = 208KB.
// ---------------------------------------------------------------------------
__global__ __launch_bounds__(256) void ffn_kernel(
    const float* __restrict__ x, const float* __restrict__ P,
    const float* __restrict__ npm,
    const float* __restrict__ ln1_g, const float* __restrict__ ln1_b,
    const float* __restrict__ W1, const float* __restrict__ b1,
    const float* __restrict__ W2, const float* __restrict__ b2,
    const float* __restrict__ ln2_g, const float* __restrict__ ln2_b,
    float* __restrict__ out)
{
  extern __shared__ unsigned smem[];
  unsigned* sW1  = smem;                     // 512 n * 32 pairs  = 16384 dw (64KB)
  unsigned* sW2  = smem + 16384;             // 64 n * 256 pairs  = 16384 dw (64KB)
  float*    sT   = (float*)(smem + 32768);   // 128 tok * 64 f32  (32KB, residual)
  unsigned* sTA  = smem + 40960;             // 128 tok * 32 pairs (16KB, A frags G1)
  unsigned* sHid = smem + 45056;             // 8 waves * 1024 dw (32KB, hid / out)

  const int t = threadIdx.x;
  const int w = t >> 5, l = t & 31;
  const int half = l >> 4, l15 = l & 15;
  const int T0 = blockIdx.x * 128;

  // stage W1 as B-fragments: sW1[n][pair p] = (W1[2p][n], W1[2p+1][n])
  // each thread: groups of (pair, 4 consecutive n) -> b128 global loads
#pragma unroll 4
  for (int i = 0; i < 16; ++i) {
    int gidx = t + i * 256;
    int p = gidx >> 7, n0 = (gidx & 127) * 4;
    float4 r0 = *(const float4*)(W1 + (size_t)(2 * p) * DI_ + n0);
    float4 r1 = *(const float4*)(W1 + (size_t)(2 * p + 1) * DI_ + n0);
    sW1[(n0 + 0) * 32 + p] = pack_bf16(r0.x, r1.x);
    sW1[(n0 + 1) * 32 + p] = pack_bf16(r0.y, r1.y);
    sW1[(n0 + 2) * 32 + p] = pack_bf16(r0.z, r1.z);
    sW1[(n0 + 3) * 32 + p] = pack_bf16(r0.w, r1.w);
  }
  // stage W2: sW2[n][pair p] = (W2[2p][n], W2[2p+1][n])
#pragma unroll 4
  for (int i = 0; i < 16; ++i) {
    int gidx = t + i * 256;
    int p = gidx >> 4, n0 = (gidx & 15) * 4;
    float4 r0 = *(const float4*)(W2 + (size_t)(2 * p) * EP_ + n0);
    float4 r1 = *(const float4*)(W2 + (size_t)(2 * p + 1) * EP_ + n0);
    sW2[(n0 + 0) * 256 + p] = pack_bf16(r0.x, r1.x);
    sW2[(n0 + 1) * 256 + p] = pack_bf16(r0.y, r1.y);
    sW2[(n0 + 2) * 256 + p] = pack_bf16(r0.z, r1.z);
    sW2[(n0 + 3) * 256 + p] = pack_bf16(r0.w, r1.w);
  }

  // stage 1: attention mixture + LN1 (2 threads per token, 32 elems each)
  {
    int tl = t >> 1, part = t & 1, e0 = part * 32;
    int T = T0 + tl;
    int b = T >> 9, s = T & 511;
    float aw[16];
#pragma unroll
    for (int h = 0; h < 16; ++h) aw[h] = P[(size_t)(b * H_ + h) * S_ + s];  // scrambled row
    float accv[32];
#pragma unroll
    for (int j = 0; j < 32; ++j) accv[j] = 0.f;
    for (int h = 0; h < 16; ++h) {
      const float4* xp = (const float4*)(x + (size_t)T * D_ + h * EP_ + e0);
      float a = aw[h];
#pragma unroll
      for (int j = 0; j < 8; ++j) {
        float4 v = xp[j];
        accv[j * 4 + 0] += a * v.x; accv[j * 4 + 1] += a * v.y;
        accv[j * 4 + 2] += a * v.z; accv[j * 4 + 3] += a * v.w;
      }
    }
    float sm = 0.f, sq = 0.f;
#pragma unroll
    for (int j = 0; j < 32; ++j) { sm += accv[j]; sq += accv[j] * accv[j]; }
    sm += __shfl_xor(sm, 1, 32);
    sq += __shfl_xor(sq, 1, 32);
    float mean = sm * (1.f / 64.f);
    float var  = sq * (1.f / 64.f) - mean * mean;
    float rstd = rsqrtf(var + 1e-5f);
    float np = npm[T];
    float tv[32];
#pragma unroll
    for (int j = 0; j < 32; ++j) {
      int e = e0 + j;
      tv[j] = ((accv[j] - mean) * rstd * ln1_g[e] + ln1_b[e]) * np;
    }
#pragma unroll
    for (int j4 = 0; j4 < 8; ++j4) {
      float4 fv; fv.x = tv[j4*4]; fv.y = tv[j4*4+1]; fv.z = tv[j4*4+2]; fv.w = tv[j4*4+3];
      *(float4*)(sT + tl * 64 + e0 + j4 * 4) = fv;           // ds_store_b128
    }
    unsigned pk[16];
#pragma unroll
    for (int jp = 0; jp < 16; ++jp) pk[jp] = pack_bf16(tv[2 * jp], tv[2 * jp + 1]);
#pragma unroll
    for (int g = 0; g < 4; ++g) {
      uint4 qq; qq.x = pk[g*4]; qq.y = pk[g*4+1]; qq.z = pk[g*4+2]; qq.w = pk[g*4+3];
      *(uint4*)(sTA + tl * 32 + part * 16 + g * 4) = qq;     // ds_store_b128
    }
  }
  __syncthreads();

  // stage 2: FFN. Each wave owns 16 tokens.
  v8f acc2[4];
#pragma unroll
  for (int et = 0; et < 4; ++et) {
    float bb = b2[et * 16 + l15];
#pragma unroll
    for (int i = 0; i < 8; ++i) acc2[et][i] = bb;
  }
  FragBF a1[2];
  {
    int tl = w * 16 + l15;
#pragma unroll
    for (int ks = 0; ks < 2; ++ks) {
      const unsigned* p = sTA + tl * 32 + ks * 16 + half * 4;
      a1[ks].q[0] = *(const uint4*)p;
      a1[ks].q[1] = *(const uint4*)(p + 8);
    }
  }
  unsigned short* hidU16 = (unsigned short*)(sHid + w * 1024); // 16 tok * 128 bf16
  const unsigned* hidU32 = sHid + w * 1024;

  for (int nc = 0; nc < 4; ++nc) {           // DI chunks of 128
    // GEMM1: hidden[nc*128 .. +128) = relu(t @ W1 + b1)
#pragma unroll
    for (int ntl = 0; ntl < 8; ++ntl) {
      int n = nc * 128 + ntl * 16 + l15;
      v8f c;
      float bb = b1[n];
#pragma unroll
      for (int i = 0; i < 8; ++i) c[i] = bb;
#pragma unroll
      for (int ks = 0; ks < 2; ++ks) {
        FragBF bfv;
        const unsigned* p = sW1 + n * 32 + ks * 16 + half * 4;
        bfv.q[0] = *(const uint4*)p;
        bfv.q[1] = *(const uint4*)(p + 8);
        c = __builtin_amdgcn_wmma_f32_16x16x32_bf16(
            false, a1[ks].v, false, bfv.v, (short)0, c, false, false);
      }
      int nl = ntl * 16 + l15;
#pragma unroll
      for (int i = 0; i < 8; ++i)
        hidU16[(i + 8 * half) * 128 + nl] = bf16_1(fmaxf(c[i], 0.f));
    }
    wave_lds_fence();  // wave-local produce->consume (wave32 lockstep)

    // GEMM2 partial: ffn += hidden_chunk @ W2_chunk
#pragma unroll
    for (int ks = 0; ks < 4; ++ks) {
      FragBF a2;
      const unsigned* p = hidU32 + l15 * 64 + ks * 16 + half * 4;
      a2.q[0] = *(const uint4*)p;
      a2.q[1] = *(const uint4*)(p + 8);
#pragma unroll
      for (int et = 0; et < 4; ++et) {
        FragBF bfv;
        const unsigned* q = sW2 + (et * 16 + l15) * 256 + nc * 64 + ks * 16 + half * 4;
        bfv.q[0] = *(const uint4*)q;
        bfv.q[1] = *(const uint4*)(q + 8);
        acc2[et] = __builtin_amdgcn_wmma_f32_16x16x32_bf16(
            false, a2.v, false, bfv.v, (short)0, acc2[et], false, false);
      }
    }
    wave_lds_fence();
  }

  // stage 3: residual + LN2 (reuse wave's hid region as f32 out staging)
  float* sOutW = (float*)(sHid + w * 1024);  // 16 tok * 64 f32
#pragma unroll
  for (int et = 0; et < 4; ++et) {
    int e = et * 16 + l15;
#pragma unroll
    for (int i = 0; i < 8; ++i) {
      int m = i + 8 * half;
      sOutW[m * 64 + e] = acc2[et][i] + sT[(w * 16 + m) * 64 + e];
    }
  }
  wave_lds_fence();
  {
    int mloc = l >> 1, part = l & 1, e0 = part * 32;
    int T = T0 + w * 16 + mloc;
    const float4* rowv = (const float4*)(sOutW + mloc * 64 + e0);
    float vbuf[32];
    float sm = 0.f, sq = 0.f;
#pragma unroll
    for (int j4 = 0; j4 < 8; ++j4) {
      float4 v = rowv[j4];
      vbuf[j4*4+0] = v.x; vbuf[j4*4+1] = v.y; vbuf[j4*4+2] = v.z; vbuf[j4*4+3] = v.w;
      sm += v.x + v.y + v.z + v.w;
      sq += v.x*v.x + v.y*v.y + v.z*v.z + v.w*v.w;
    }
    sm += __shfl_xor(sm, 1, 32);
    sq += __shfl_xor(sq, 1, 32);
    float mean = sm * (1.f / 64.f);
    float var  = sq * (1.f / 64.f) - mean * mean;
    float rstd = rsqrtf(var + 1e-5f);
    float np = npm[T];
#pragma unroll
    for (int j4 = 0; j4 < 8; ++j4) {
      float4 ov;
      int e = e0 + j4 * 4;
      ov.x = ((vbuf[j4*4+0] - mean) * rstd * ln2_g[e+0] + ln2_b[e+0]) * np;
      ov.y = ((vbuf[j4*4+1] - mean) * rstd * ln2_g[e+1] + ln2_b[e+1]) * np;
      ov.z = ((vbuf[j4*4+2] - mean) * rstd * ln2_g[e+2] + ln2_b[e+2]) * np;
      ov.w = ((vbuf[j4*4+3] - mean) * rstd * ln2_g[e+3] + ln2_b[e+3]) * np;
      *(float4*)(out + (size_t)T * 64 + e) = ov;             // global_store_b128
    }
  }
}

// ---------------------------------------------------------------------------
extern "C" void kernel_launch(void* const* d_in, const int* in_sizes, int n_in,
                              void* d_out, int out_size, void* d_ws, size_t ws_size,
                              hipStream_t stream) {
  (void)in_sizes; (void)n_in; (void)out_size; (void)ws_size;
  const float*         x    = (const float*)d_in[0];
  const unsigned char* mask = (const unsigned char*)d_in[1];  // jnp bool_ = 1 byte
  const float*         npm  = (const float*)d_in[2];
  const float*         Q    = (const float*)d_in[3];
  const float*         Wk   = (const float*)d_in[4];
  const float*         bk   = (const float*)d_in[5];
  const float*         g1   = (const float*)d_in[6];
  const float*         bb1  = (const float*)d_in[7];
  const float*         W1   = (const float*)d_in[8];
  const float*         b1   = (const float*)d_in[9];
  const float*         W2   = (const float*)d_in[10];
  const float*         b2   = (const float*)d_in[11];
  const float*         g2   = (const float*)d_in[12];
  const float*         bb2  = (const float*)d_in[13];

  float* out    = (float*)d_out;                      // (B,S,EP)
  float* attn_w = out + (size_t)B_ * S_ * EP_;        // (B,H,S) flat == P flat
  float* scores = (float*)d_ws;                       // 2048 * 512 f32 (4MB)

  kproj_scores_kernel<<<dim3(512), dim3(256), 96 * 1024, stream>>>(x, Wk, bk, Q, scores);
  softmax_kernel<<<dim3(2048), dim3(256), 0, stream>>>(scores, mask, attn_w);
  ffn_kernel<<<dim3(512), dim3(256), 208 * 1024, stream>>>(
      x, scores, npm, g1, bb1, W1, b1, W2, b2, g2, bb2, out);
}
